// MultiHeadedAttention_67242007987141
// MI455X (gfx1250) — matrix operand order repaired and checked by reading, not compiled, not run on hardware
//
#include <hip/hip_runtime.h>
#include <cstdint>
#include <cstddef>

#define B_  2
#define L_  2048
#define D_  1024
#define H_  16
#define DK_ 64

typedef __attribute__((ext_vector_type(16))) __bf16 bf16x16;
typedef __attribute__((ext_vector_type(8)))  __bf16 bf16x8;
typedef __attribute__((ext_vector_type(4)))  __bf16 bf16x4;
typedef __attribute__((ext_vector_type(8)))  float  f32x8;
typedef __attribute__((ext_vector_type(4)))  float  f32x4;

static __device__ __forceinline__ f32x8 wmma_bf16(bf16x16 a, bf16x16 b, f32x8 c) {
    // D(16x16,f32) = A(16x32,bf16) x B(32x16,bf16) + C
    return __builtin_amdgcn_wmma_f32_16x16x32_bf16(false, a, false, b, (short)0, c, false, false);
}

// Build a 16-element fragment from two aligned 8-element LDS chunks.
static __device__ __forceinline__ bf16x16 frag_from2(const __bf16* lo, const __bf16* hi) {
    bf16x8 a = *(const bf16x8*)lo;
    bf16x8 b = *(const bf16x8*)hi;
    bf16x16 r;
    for (int i = 0; i < 8; ++i) { r[i] = a[i]; r[i + 8] = b[i]; }
    return r;
}

// ---------------------------------------------------------------------------
// Kernel 1: projection GEMM  Y = X(4096x1024) @ W(1024x1024), emit bf16.
//   TR==0 : dst[(b*H+h)*L + l][dk]  (Q_up / K_up layout)
//   TR==1 : dst[(b*H+h)*DK + dk][l] (V_up^T layout, contiguous over l)
// Workgroup: 128 threads (4 waves), tile 64x64, K-step 32. LDS rows padded to
// 40 bf16 (80B): 16B-aligned vector chunks + conflict-free half-wave reads.
// ---------------------------------------------------------------------------
#define APAD 40
template <int TR>
__global__ __launch_bounds__(128)
void proj_kernel(const float* __restrict__ X, const float* __restrict__ W,
                 __bf16* __restrict__ dst) {
    __shared__ __bf16 As[64 * APAD];   // [m][k]
    __shared__ __bf16 Bs[64 * APAD];   // [n][k] (transposed W tile)

    const int tid  = threadIdx.x;
    const int wave = tid >> 5;
    const int lane = tid & 31;
    const int gm0  = blockIdx.y * 64;   // token-block
    const int gn0  = blockIdx.x * 64;   // output-feature block

    f32x8 acc[4];
    for (int nt = 0; nt < 4; ++nt)
        for (int e = 0; e < 8; ++e) acc[nt][e] = 0.f;

    const int row = wave * 16 + (lane & 15);  // local A row for this lane
    const int kg  = lane >> 4;                // half-wave K group
    const int l15 = lane & 15;

    // staging coordinates
    const int am   = tid >> 1;          // A: row 0..63
    const int ah   = (tid & 1) * 16;    // A: 16-float half of the 32-wide k
    const int bk   = tid >> 2;          // B: k 0..31
    const int bn   = (tid & 3) * 16;    // B: 16-col group

    for (int kb = 0; kb < 1024; kb += 32) {
        // ---- stage A: 4x float4 loads -> two 16B bf16 vector stores -------
        {
            const f32x4* xp = (const f32x4*)(X + (size_t)(gm0 + am) * 1024 + kb + ah);
            f32x4 x0 = xp[0], x1 = xp[1], x2 = xp[2], x3 = xp[3];
            bf16x8 o0, o1;
            o0[0]=(__bf16)x0.x; o0[1]=(__bf16)x0.y; o0[2]=(__bf16)x0.z; o0[3]=(__bf16)x0.w;
            o0[4]=(__bf16)x1.x; o0[5]=(__bf16)x1.y; o0[6]=(__bf16)x1.z; o0[7]=(__bf16)x1.w;
            o1[0]=(__bf16)x2.x; o1[1]=(__bf16)x2.y; o1[2]=(__bf16)x2.z; o1[3]=(__bf16)x2.w;
            o1[4]=(__bf16)x3.x; o1[5]=(__bf16)x3.y; o1[6]=(__bf16)x3.z; o1[7]=(__bf16)x3.w;
            *(bf16x8*)(As + am * APAD + ah)     = o0;
            *(bf16x8*)(As + am * APAD + ah + 8) = o1;
        }
        // ---- stage B transposed: 4x float4 loads -> 16 b16 LDS stores -----
        {
            const f32x4* wp = (const f32x4*)(W + (size_t)(kb + bk) * 1024 + gn0 + bn);
            for (int c = 0; c < 4; ++c) {
                f32x4 w4 = wp[c];
                Bs[(bn + c * 4 + 0) * APAD + bk] = (__bf16)w4.x;
                Bs[(bn + c * 4 + 1) * APAD + bk] = (__bf16)w4.y;
                Bs[(bn + c * 4 + 2) * APAD + bk] = (__bf16)w4.z;
                Bs[(bn + c * 4 + 3) * APAD + bk] = (__bf16)w4.w;
            }
        }
        __syncthreads();

        // A fragment: lane(0-15)=row M; i<8 -> K=i+8kg, i>=8 -> K=i+8+8kg
        bf16x16 a = frag_from2(As + row * APAD + 8 * kg,
                               As + row * APAD + 16 + 8 * kg);
        for (int nt = 0; nt < 4; ++nt) {
            // B fragment: lane holds column N; i<8 -> K=i+8kg, i>=8 -> K=i+8+8kg
            const int col = nt * 16 + l15;
            bf16x16 b = frag_from2(Bs + col * APAD + 8 * kg,
                                   Bs + col * APAD + 16 + 8 * kg);
            acc[nt] = wmma_bf16(a, b, acc[nt]);
        }
        __syncthreads();
    }

    // C layout: VGPR r -> M = r + 8*(lane>=16), N = lane&15
    for (int nt = 0; nt < 4; ++nt) {
        for (int vg = 0; vg < 8; ++vg) {
            const int M  = vg + 8 * kg;
            const int gm = gm0 + wave * 16 + M;          // token = b*L + l
            const int gn = gn0 + nt * 16 + l15;          // d = h*64 + dk
            const int b  = gm >> 11, l = gm & (L_ - 1);
            const int h  = gn >> 6,  dk = gn & (DK_ - 1);
            const __bf16 v = (__bf16)acc[nt][vg];
            if (TR == 0)
                dst[(((size_t)(b * H_ + h)) * L_ + l) * DK_ + dk] = v;
            else
                dst[(((size_t)(b * H_ + h)) * DK_ + dk) * L_ + l] = v;
        }
    }
}

// ---------------------------------------------------------------------------
// Kernel 2: fused  S = QK^T/8, mask, softmax (in 320KB LDS), attn store, P@V.
// One workgroup (8 waves) per (bh, 16-query stripe). Padded LDS rows:
//   sS: 2052 f32/row, sP: 2056 bf16/row, sQ: 72 bf16/row (bank-conflict free,
//   16B-aligned so softmax runs on f32x4 LDS vectors).
// ---------------------------------------------------------------------------
#define SP_S 2052
#define SP_P 2056
#define SP_Q 72
#define SMEM_S   (16 * SP_S * 4)
#define SMEM_P   (16 * SP_P * 2)
#define SMEM_Q   (16 * SP_Q * 2)
#define SMEM_O   (16 * 64 * 4)
#define SMEM_R   (256 * 4)
#define ATTN_SMEM_BYTES (SMEM_S + SMEM_P + SMEM_Q + SMEM_O + SMEM_R)

__global__ __launch_bounds__(256)
void attn_kernel(const __bf16* __restrict__ q_up, const __bf16* __restrict__ k_up,
                 const __bf16* __restrict__ vT, const unsigned char* __restrict__ mask,
                 float* __restrict__ attn_out, float* __restrict__ out_ws) {
    extern __shared__ char smem[];
    float*  sS = (float*)smem;
    __bf16* sP = (__bf16*)(smem + SMEM_S);
    __bf16* sQ = (__bf16*)(smem + SMEM_S + SMEM_P);
    float*  sO = (float*)(smem + SMEM_S + SMEM_P + SMEM_Q);
    float*  sR = (float*)(smem + SMEM_S + SMEM_P + SMEM_Q + SMEM_O);

    const int tid  = threadIdx.x;
    const int wave = tid >> 5;
    const int lane = tid & 31;
    const int q0   = blockIdx.x * 16;
    const int bh   = blockIdx.y;
    const int b    = bh / H_;
    const int h    = bh % H_;
    const int kg   = lane >> 4;
    const int l15  = lane & 15;

    // stage Q tile (vectorized 8B chunks) and zero the output accumulator
    {
        const int idx = tid * 4;                  // 1024 bf16 / 256 thr
        const int r = idx >> 6, d = idx & 63;
        *(uint64_t*)(sQ + r * SP_Q + d) =
            *(const uint64_t*)(q_up + ((size_t)bh * L_ + q0 + r) * DK_ + d);
    }
    for (int idx = tid; idx < 16 * 64; idx += 256) sO[idx] = 0.f;
    __syncthreads();

    // ---- scores: each wave covers 16 of the 128 key tiles -----------------
    bf16x16 aQ[2];
    for (int ks = 0; ks < 2; ++ks)
        aQ[ks] = frag_from2(sQ + l15 * SP_Q + ks * 32 + 16 * kg,
                            sQ + l15 * SP_Q + ks * 32 + 16 * kg + 8);

    for (int nt = wave; nt < 128; nt += 8) {
        f32x8 c;
        for (int e = 0; e < 8; ++e) c[e] = 0.f;
        const __bf16* kbase = k_up + ((size_t)bh * L_ + nt * 16 + l15) * DK_ + kg * 16;
        __builtin_prefetch((const void*)(kbase + 8 * 16 * DK_), 0, 3); // next tile
        for (int ks = 0; ks < 2; ++ks) {
            bf16x16 bfr = *(const bf16x16*)(kbase + ks * 32);  // 32B contiguous
            c = wmma_bf16(aQ[ks], bfr, c);
        }
        for (int vg = 0; vg < 8; ++vg) {
            const int M = vg + 8 * kg;          // query row in stripe
            const int k = nt * 16 + l15;        // key index
            sS[M * SP_S + k] = c[vg] * 0.125f;  // 1/sqrt(64); mask applied later
        }
    }
    __syncthreads();

    // ---- mask + softmax: row = tid&15 (conflict-free), 128-elem chunks,
    //      f32x4-vectorized LDS + global streams ----------------------------
    {
        const int row  = tid & 15;
        const int c16  = tid >> 4;
        const int ck   = c16 * 128;
        float* srow = sS + row * SP_S + ck;
        const unsigned int* mrow = (const unsigned int*)
            (mask + ((size_t)b * L_ + q0 + row) * L_ + ck);

        float mx = -3.0e38f;
        for (int w = 0; w < 32; ++w) {          // 4 scores + 4 mask bytes/iter
            unsigned int u = mrow[w];
            f32x4 s4 = *(f32x4*)(srow + w * 4);
            if ( u        & 0xffu) s4.x = -1e9f;
            if ((u >>  8) & 0xffu) s4.y = -1e9f;
            if ((u >> 16) & 0xffu) s4.z = -1e9f;
            if ((u >> 24) & 0xffu) s4.w = -1e9f;
            *(f32x4*)(srow + w * 4) = s4;
            mx = fmaxf(fmaxf(fmaxf(mx, s4.x), fmaxf(s4.y, s4.z)), s4.w);
        }
        sR[c16 * 16 + row] = mx;
        __syncthreads();
        float rmax = sR[row];
        for (int j = 1; j < 16; ++j) rmax = fmaxf(rmax, sR[j * 16 + row]);
        __syncthreads();

        float lsum = 0.f;
        for (int w = 0; w < 32; ++w) {
            f32x4 s4 = *(f32x4*)(srow + w * 4);
            s4.x = __expf(s4.x - rmax);
            s4.y = __expf(s4.y - rmax);
            s4.z = __expf(s4.z - rmax);
            s4.w = __expf(s4.w - rmax);
            *(f32x4*)(srow + w * 4) = s4;
            lsum += (s4.x + s4.y) + (s4.z + s4.w);
        }
        sR[c16 * 16 + row] = lsum;
        __syncthreads();
        float rsum = 0.f;
        for (int j = 0; j < 16; ++j) rsum += sR[j * 16 + row];
        const float inv = 1.f / rsum;

        float*  arow = attn_out + (size_t)bh * L_ * L_ + (size_t)(q0 + row) * L_ + ck;
        __bf16* prow = sP + row * SP_P + ck;
        for (int w = 0; w < 32; ++w) {
            f32x4 p4 = *(f32x4*)(srow + w * 4) * inv;
            *(f32x4*)(arow + w * 4) = p4;       // attn: b128 stores, written once
            bf16x4 pb;
            pb[0] = (__bf16)p4.x; pb[1] = (__bf16)p4.y;
            pb[2] = (__bf16)p4.z; pb[3] = (__bf16)p4.w;
            *(bf16x4*)(prow + w * 4) = pb;      // bf16 copy for P@V
        }
    }
    __syncthreads();

    // ---- P @ V: waves split 4 dk-tiles x 2 key-halves ---------------------
    {
        const int nt    = wave & 3;
        const int khalf = wave >> 2;
        const int col   = nt * 16 + l15;
        f32x8 c;
        for (int e = 0; e < 8; ++e) c[e] = 0.f;
        const __bf16* vbase = vT + ((size_t)bh * DK_ + col) * L_ + khalf * 1024 + kg * 16;
        const __bf16* prow  = sP + l15 * SP_P + khalf * 1024;
        for (int ks = 0; ks < 32; ++ks) {
            __builtin_prefetch((const void*)(vbase + (ks + 4) * 32), 0, 3);
            bf16x16 a = frag_from2(prow + ks * 32 + 16 * kg,
                                   prow + ks * 32 + 16 * kg + 8);
            bf16x16 bfr = *(const bf16x16*)(vbase + ks * 32);
            c = wmma_bf16(a, bfr, c);
        }
        for (int vg = 0; vg < 8; ++vg) {
            const int M = vg + 8 * kg;
            atomicAdd(&sO[M * 64 + nt * 16 + l15], c[vg]);
        }
    }
    __syncthreads();

    for (int idx = tid; idx < 16 * 64; idx += 256) {
        int r = idx >> 6, d = idx & 63;
        out_ws[((size_t)b * L_ + q0 + r) * D_ + h * 64 + d] = sO[idx];
    }
}

// ---------------------------------------------------------------------------
// Kernel 3: residual + LayerNorm, one block per token, float4 I/O.
// ---------------------------------------------------------------------------
__global__ __launch_bounds__(256)
void ln_kernel(const float* __restrict__ out_ws, const float* __restrict__ query,
               const float* __restrict__ g, const float* __restrict__ bta,
               float* __restrict__ normed) {
    __shared__ float red[256];
    const int t   = blockIdx.x;
    const int tid = threadIdx.x;
    const size_t base = (size_t)t * D_;
    const int i0 = tid * 4;

    f32x4 v = *(const f32x4*)(out_ws + base + i0);
    f32x4 q = *(const f32x4*)(query  + base + i0);
    v = v + q;
    float s = v.x + v.y + v.z + v.w;
    red[tid] = s;
    __syncthreads();
    for (int off = 128; off > 0; off >>= 1) {
        if (tid < off) red[tid] += red[tid + off];
        __syncthreads();
    }
    const float mu = red[0] * (1.f / 1024.f);
    __syncthreads();

    f32x4 d = v - mu;
    red[tid] = d.x * d.x + d.y * d.y + d.z * d.z + d.w * d.w;
    __syncthreads();
    for (int off = 128; off > 0; off >>= 1) {
        if (tid < off) red[tid] += red[tid + off];
        __syncthreads();
    }
    const float var = red[0] * (1.f / 1024.f);
    const float rs  = rsqrtf(var + 1e-6f);

    f32x4 gg = *(const f32x4*)(g   + i0);
    f32x4 bb = *(const f32x4*)(bta + i0);
    f32x4 o  = d * rs * gg + bb;
    *(f32x4*)(normed + base + i0) = o;
}

// ---------------------------------------------------------------------------
extern "C" void kernel_launch(void* const* d_in, const int* in_sizes, int n_in,
                              void* d_out, int out_size, void* d_ws, size_t ws_size,
                              hipStream_t stream) {
    const float* key   = (const float*)d_in[0];
    const float* value = (const float*)d_in[1];
    const float* query = (const float*)d_in[2];
    const unsigned char* mask = (const unsigned char*)d_in[3];
    const float* Wk   = (const float*)d_in[4];
    const float* Wv   = (const float*)d_in[5];
    const float* Wq   = (const float*)d_in[6];
    const float* ln_g = (const float*)d_in[7];
    const float* ln_b = (const float*)d_in[8];

    float* normed = (float*)d_out;
    float* attn   = (float*)d_out + (size_t)B_ * L_ * D_;

    char* ws = (char*)d_ws;
    __bf16* k_up = (__bf16*)(ws);                       // 8 MB
    __bf16* vT   = (__bf16*)(ws + (8ull << 20));        // 8 MB (transposed layout)
    __bf16* q_up = (__bf16*)(ws + (16ull << 20));       // 8 MB
    float*  outb = (float*)(ws + (24ull << 20));        // 16 MB pre-LN output

    // 1) projections: K, V (transposed), Q
    dim3 pgrid(D_ / 64, (B_ * L_) / 64);
    proj_kernel<0><<<pgrid, 128, 0, stream>>>(key,   Wk, k_up);
    proj_kernel<1><<<pgrid, 128, 0, stream>>>(value, Wv, vT);
    proj_kernel<0><<<pgrid, 128, 0, stream>>>(query, Wq, q_up);

    // 2) fused attention (needs >64KB dynamic LDS; CDNA5 WGP has 320KB)
    hipFuncSetAttribute(reinterpret_cast<const void*>(attn_kernel),
                        hipFuncAttributeMaxDynamicSharedMemorySize,
                        ATTN_SMEM_BYTES);
    dim3 agrid(L_ / 16, B_ * H_);
    attn_kernel<<<agrid, 256, ATTN_SMEM_BYTES, stream>>>(q_up, k_up, vT, mask, attn, outb);

    // 3) residual + LayerNorm
    ln_kernel<<<B_ * L_, 256, 0, stream>>>(outb, query, ln_g, ln_b, normed);
}
